// LFADS_65111704207494
// MI455X (gfx1250) — compile-verified, matrix operand order checked by reading, not run on hardware
//
#include <hip/hip_runtime.h>
#include <math.h>

// LFADS dims
#define B_   256
#define T_   400
#define N_   128
#define O_   16
#define H_   64
#define F_   64
#define NI   144      // N+O
#define NIP  160      // padded to multiple of 32 for K-steps
#define G3   192      // 3*H
#define CLIPV 200.0f
#define KAPPA 0.1f

typedef __attribute__((ext_vector_type(16))) _Float16 v16h;
typedef __attribute__((ext_vector_type(8)))  float    v8f;
typedef __attribute__((ext_vector_type(4)))  unsigned int v4u;
typedef __attribute__((ext_vector_type(8)))  int v8i_;
typedef __attribute__((ext_vector_type(4)))  int v4i_;

#if __has_builtin(__builtin_amdgcn_tensor_load_to_lds) && \
    __has_builtin(__builtin_amdgcn_s_wait_tensorcnt)
#define USE_TDM 1
#else
#define USE_TDM 0
#endif

// --- WMMA helpers ------------------------------------------------------------
// 16-bit A (16xK) / B (Kx16) fragment element -> K offset mapping (ISA 7.12.2):
// lane = row%16 (A: M, B: N); lanes 0-15 hold K {0..7,16..23}, lanes 16-31 hold
// K {8..15,24..31} within each 32-wide K step, 2 halves per VGPR.
__device__ __forceinline__ int kmap16(int e, int hi) {
  int v = e >> 1;
  return (v < 4 ? (v << 1) : 16 + ((v - 4) << 1)) + (e & 1) + (hi << 3);
}

__device__ __forceinline__ v16h frag_lds(const _Float16* base, int row, int stride,
                                         int k0, int hi) {
  v16h f;
#pragma unroll
  for (int e = 0; e < 16; ++e) f[e] = base[row * stride + k0 + kmap16(e, hi)];
  return f;
}

__device__ __forceinline__ v8f wmma32(v16h a, v16h b, v8f c) {
  return __builtin_amdgcn_wmma_f32_16x16x32_f16(false, a, false, b, (short)0, c,
                                                false, false);
}

__device__ __forceinline__ float sigm(float x) { return 1.f / (1.f + expf(-x)); }
__device__ __forceinline__ float clipv(float x) {
  return fminf(fmaxf(x, -CLIPV), CLIPV);
}

// --- Tensor Data Mover: 1-D contiguous global->LDS copy (D# per ISA 8.x) -----
__device__ __forceinline__ unsigned ldsoff(const void* p) {
  return (unsigned)(unsigned long long)p;  // LDS aperture: offset in low 32b
}

#if USE_TDM
__device__ __forceinline__ void tdm_load_1d(unsigned lds_addr,
                                            unsigned long long gaddr,
                                            unsigned ndw) {
  v4u g0;
  g0[0] = 1u;                                        // count=1, user mode
  g0[1] = lds_addr;                                  // lds_addr (bytes)
  g0[2] = (unsigned)(gaddr & 0xffffffffull);         // global_addr[31:0]
  g0[3] = (unsigned)((gaddr >> 32) & 0x01ffffffull)  // global_addr[56:32]
          | (2u << 30);                              // type=2 ("image")
  v8i_ g1;
  g1[0] = (int)(2u << 16);                 // workgroup_mask=0, data_size=4B
  g1[1] = (int)((ndw & 0xffffu) << 16);    // tensor_dim0[15:0]
  g1[2] = (int)((ndw >> 16) & 0xffffu) | (1 << 16);  // td0 hi | tensor_dim1=1
  g1[3] = (int)(ndw << 16);                // td1 hi=0 | tile_dim0=ndw
  g1[4] = 1;                               // tile_dim1=1, tile_dim2=0
  g1[5] = (int)ndw;                        // tensor_dim0_stride[31:0]
  g1[6] = 0;
  g1[7] = 0;
  v4i_ gz = {0, 0, 0, 0};
#if defined(__clang_major__) && __clang_major__ >= 23
  v8i_ gz8 = {0, 0, 0, 0, 0, 0, 0, 0};
  __builtin_amdgcn_tensor_load_to_lds(g0, g1, gz, gz, gz8, 0);
#else
  __builtin_amdgcn_tensor_load_to_lds(g0, g1, gz, gz, 0);
#endif
}
#endif

// --- Kernel 1: pack xa = concat(x,a).transpose(2,0,1)+1e-8 as f16 (T,B,160) ---
__global__ void pack_xa_kernel(const float* __restrict__ x,
                               const float* __restrict__ a,
                               _Float16* __restrict__ xa) {
  size_t idx = (size_t)blockIdx.x * 256 + threadIdx.x;
  if (idx >= (size_t)T_ * B_ * NIP) return;
  int i = (int)(idx % NIP);
  size_t tb = idx / NIP;
  int b = (int)(tb % B_);
  int t = (int)(tb / B_);
  float v = 0.f;
  if (i < N_)       v = x[((size_t)b * N_ + i) * T_ + t] + 1e-8f;
  else if (i < NI)  v = a[((size_t)b * O_ + (i - N_)) * T_ + t] + 1e-8f;
  xa[idx] = (_Float16)v;
}

// --- Kernel 2: the four sequence GRUs (enc fw/bw, con_bf fw/bw) --------------
struct ScanArgs {
  const float* Wih[4]; const float* Whh[4];
  const float* bih[4]; const float* bhh[4];
  const _Float16* xa;      // (T,B,NIP) f16
  _Float16* econ;          // (T,B,128) f16: cols 0-63 fw, 64-127 bw
  float* hfw; float* hbw;  // (B,64)
};

__global__ __launch_bounds__(128) void gru_scan_kernel(ScanArgs sa) {
  const int gru = blockIdx.y;     // 0 enc_fw, 1 enc_bw, 2 con_fw, 3 con_bw
  const int bt  = blockIdx.x;     // batch tile (16 rows)
  const int tid = threadIdx.x, lane = tid & 31, wave = tid >> 5;
  const int lr = lane & 15, lhi = lane >> 4;

  extern __shared__ char smem[];
  _Float16* wih  = (_Float16*)smem;           // [192][160]
  _Float16* whh  = wih + G3 * NIP;            // [192][64]
  _Float16* axa0 = whh + G3 * H_;             // [2][16][160] double buffer
  _Float16* h16  = axa0 + 2 * 16 * NIP;       // [16][64]
  float* h32  = (float*)(h16 + 16 * H_);      // [16][64]
  float* xpT  = h32 + 16 * H_;                // [192][16]  transposed
  float* hpT  = xpT + G3 * 16;                // [192][16]
  float* bihL = hpT + G3 * 16;                // [192]
  float* bhhL = bihL + G3;                    // [192]

  const float* Wih = sa.Wih[gru]; const float* Whh = sa.Whh[gru];
  const float* bih = sa.bih[gru]; const float* bhh = sa.bhh[gru];

  for (int i = tid; i < G3 * NIP; i += 128) {
    int n = i / NIP, k = i - n * NIP;
    wih[i] = (_Float16)(k < NI ? Wih[n * NI + k] : 0.f);
  }
  for (int i = tid; i < G3 * H_; i += 128) whh[i] = (_Float16)Whh[i];
  for (int i = tid; i < G3; i += 128) { bihL[i] = bih[i]; bhhL[i] = bhh[i]; }
  for (int i = tid; i < 16 * H_; i += 128) { h32[i] = 0.f; h16[i] = (_Float16)0.f; }
  __syncthreads();

  const bool rev = (gru == 1) || (gru == 3);
  const unsigned long long xabase = (unsigned long long)sa.xa;
  const unsigned axaoff = ldsoff(axa0);
#if USE_TDM
  if (wave == 0) {
    const int t0 = rev ? (T_ - 1) : 0;
    tdm_load_1d(axaoff, xabase + ((size_t)t0 * B_ + bt * 16) * NIP * 2ull,
                16 * NIP / 2);
  }
#endif
  for (int step = 0; step < T_; ++step) {
    const int t = rev ? (T_ - 1 - step) : step;
    const int cur = step & 1;
    // branchless LDS buffer select: keeps addrspace(3) provenance (ds_load)
    const _Float16* axa = axa0 + cur * (16 * NIP);
#if USE_TDM
    if (wave == 0) __builtin_amdgcn_s_wait_tensorcnt(0);
    __syncthreads();
    if (wave == 0 && step + 1 < T_) {
      const int tn = rev ? (T_ - 2 - step) : (step + 1);
      tdm_load_1d(axaoff + (unsigned)(cur ^ 1) * (16 * NIP * 2),
                  xabase + ((size_t)tn * B_ + bt * 16) * NIP * 2ull, 16 * NIP / 2);
    }
#else
    {
      const unsigned* src =
          (const unsigned*)(sa.xa + ((size_t)t * B_ + bt * 16) * NIP);
      unsigned* dstp = (unsigned*)(axa0 + cur * (16 * NIP));
      for (int i = tid; i < 16 * NIP / 2; i += 128) dstp[i] = src[i];
      __syncthreads();
    }
#endif

    // xp = xa@Wih^T (K=160), hp = h@Whh^T (K=64); 12 N-tiles / 4 waves
#pragma unroll
    for (int q = 0; q < 3; ++q) {
      const int nt = wave * 3 + q;
      v8f xacc = {}; v8f hacc = {};
#pragma unroll
      for (int ks = 0; ks < 5; ++ks) {
        v16h af = frag_lds(axa, lr, NIP, ks * 32, lhi);
        v16h bf = frag_lds(wih, nt * 16 + lr, NIP, ks * 32, lhi);
        xacc = wmma32(af, bf, xacc);
      }
#pragma unroll
      for (int ks = 0; ks < 2; ++ks) {
        v16h af = frag_lds(h16, lr, H_, ks * 32, lhi);
        v16h bf = frag_lds(whh, nt * 16 + lr, H_, ks * 32, lhi);
        hacc = wmma32(af, bf, hacc);
      }
      const int n = nt * 16 + lr;
      const float bi = bihL[n], bh = bhhL[n];
      float* xd = &xpT[n * 16 + lhi * 8];
      float* hd = &hpT[n * 16 + lhi * 8];
#pragma unroll
      for (int r8 = 0; r8 < 8; ++r8) { xd[r8] = xacc[r8] + bi; hd[r8] = hacc[r8] + bh; }
    }
    __syncthreads();

    for (int i = tid; i < 16 * H_; i += 128) {
      const int m = i >> 6, h = i & 63;
      float rg = sigm(xpT[h * 16 + m] + hpT[h * 16 + m]);
      float zg = sigm(xpT[(64 + h) * 16 + m] + hpT[(64 + h) * 16 + m]);
      float ng = tanhf(xpT[(128 + h) * 16 + m] + rg * hpT[(128 + h) * 16 + m]);
      float hn = clipv((1.f - zg) * ng + zg * h32[i]);
      h32[i] = hn; h16[i] = (_Float16)hn;
      if (gru >= 2)
        sa.econ[((size_t)t * B_ + bt * 16 + m) * 128 + (gru == 2 ? 0 : 64) + h] =
            (_Float16)hn;
    }
    __syncthreads();
  }
  if (gru < 2) {
    float* dstf = (gru == 0) ? sa.hfw : sa.hbw;
    for (int i = tid; i < 16 * H_; i += 128)
      dstf[(bt * 16 + (i >> 6)) * H_ + (i & 63)] = h32[i];
  }
}

// --- Kernel 3: g0 sample -----------------------------------------------------
__global__ void g0_kernel(const float* hfw, const float* hbw,
                          const float* Wmu, const float* bmu,
                          const float* Wsig, const float* bsig,
                          const float* eps, float* g0out) {
  const int b = blockIdx.x, h = threadIdx.x;  // 64 threads
  float mu = bmu[h], sg = bsig[h];
  for (int i = 0; i < 2 * H_; ++i) {
    float e = (i < H_) ? hfw[b * H_ + i] : hbw[b * H_ + i - H_];
    mu += e * Wmu[h * 2 * H_ + i];
    sg += e * Wsig[h * 2 * H_ + i];
  }
  g0out[b * H_ + h] = eps[b * H_ + h] * expf(sg * 0.5f) + mu;
}

// --- Kernel 4: fused decoder scan -------------------------------------------
struct DecArgs {
  const float *cfWih, *cfWhh, *cfbih, *cfbhh;
  const float *gWih, *gWhh, *gbih, *gbhh;
  const float *Wmu, *bmu, *Wsig, *bsig;
  const float *Wfac, *bfac, *Wrate, *brate;
  const float *c0, *f0, *g0, *epsu;
  const _Float16* econ;
  float *rates, *factors, *states, *inferred;  // (B,*,T)
};

__global__ __launch_bounds__(128) void dec_kernel(DecArgs da) {
  const int bt = blockIdx.x;
  const int tid = threadIdx.x, lane = tid & 31, wave = tid >> 5;
  const int lr = lane & 15, lhi = lane >> 4;

  extern __shared__ char smem[];
  _Float16* cfwih = (_Float16*)smem;          // [192][192]
  _Float16* cfwhh = cfwih + G3 * G3;          // [192][64]
  _Float16* gwih  = cfwhh + G3 * H_;          // [192][64]
  _Float16* gwhh  = gwih + G3 * H_;           // [192][64]
  _Float16* wmu   = gwhh + G3 * H_;           // [64][64]
  _Float16* wsig  = wmu + H_ * H_;            // [64][64]
  _Float16* wfac  = wsig + H_ * H_;           // [64][64]
  _Float16* wrate = wfac + H_ * H_;           // [128][64]
  _Float16* ebuf0 = wrate + N_ * H_;          // [2][16][128] double buffer
  _Float16* fbuf  = ebuf0 + 2 * 16 * 128;     // [16][64]
  _Float16* c16   = fbuf + 16 * H_;           // [16][64]
  _Float16* g16   = c16 + 16 * H_;            // [16][64]
  _Float16* u16   = g16 + 16 * H_;            // [16][64]
  float* c32    = (float*)(u16 + 16 * H_);    // [16][64]
  float* g32    = c32 + 16 * H_;              // [16][64]
  float* xpT    = g32 + 16 * H_;              // [192][16] transposed
  float* hpT    = xpT + G3 * 16;              // [192][16]
  float* musigT = hpT + G3 * 16;              // [128][16] transposed
  float* cfbihL = musigT + 2 * H_ * 16;       // 192
  float* cfbhhL = cfbihL + G3;                // 192
  float* gbihL  = cfbhhL + G3;                // 192
  float* gbhhL  = gbihL + G3;                 // 192
  float* bmuL   = gbhhL + G3;                 // 64
  float* bsigL  = bmuL + H_;                  // 64
  float* bfacL  = bsigL + H_;                 // 64
  float* brateL = bfacL + H_;                 // 128

  for (int i = tid; i < G3 * G3; i += 128) cfwih[i] = (_Float16)da.cfWih[i];
  for (int i = tid; i < G3 * H_; i += 128) {
    cfwhh[i] = (_Float16)da.cfWhh[i];
    gwih[i]  = (_Float16)da.gWih[i];
    gwhh[i]  = (_Float16)da.gWhh[i];
  }
  for (int i = tid; i < H_ * H_; i += 128) {
    wmu[i]  = (_Float16)da.Wmu[i];
    wsig[i] = (_Float16)da.Wsig[i];
    wfac[i] = (_Float16)da.Wfac[i];
  }
  for (int i = tid; i < N_ * H_; i += 128) wrate[i] = (_Float16)da.Wrate[i];
  for (int i = tid; i < G3; i += 128) {
    cfbihL[i] = da.cfbih[i]; cfbhhL[i] = da.cfbhh[i];
    gbihL[i]  = da.gbih[i];  gbhhL[i]  = da.gbhh[i];
  }
  for (int i = tid; i < H_; i += 128) {
    bmuL[i] = da.bmu[i]; bsigL[i] = da.bsig[i]; bfacL[i] = da.bfac[i];
  }
  for (int i = tid; i < N_; i += 128) brateL[i] = da.brate[i];
  // init carries
  for (int i = tid; i < 16 * H_; i += 128) {
    const int m = i >> 6, h = i & 63;
    float cv = da.c0[h];
    float gv = da.g0[(bt * 16 + m) * H_ + h];
    c32[i] = cv; c16[i] = (_Float16)cv;
    g32[i] = gv; g16[i] = (_Float16)gv;
    fbuf[m * H_ + h] = (_Float16)da.f0[h];
  }
  __syncthreads();

  const unsigned long long ebase = (unsigned long long)da.econ;
  const unsigned eoff = ldsoff(ebuf0);
#if USE_TDM
  if (wave == 0)
    tdm_load_1d(eoff, ebase + ((size_t)0 * B_ + bt * 16) * 256ull, 1024);
#endif
  for (int t = 0; t < T_; ++t) {
    const int cur = t & 1;
    const _Float16* ebuf = ebuf0 + cur * (16 * 128);  // branchless LDS select
#if USE_TDM
    if (wave == 0) __builtin_amdgcn_s_wait_tensorcnt(0);
    __syncthreads();
    if (wave == 0 && t + 1 < T_)
      tdm_load_1d(eoff + (unsigned)(cur ^ 1) * (16 * 128 * 2),
                  ebase + ((size_t)(t + 1) * B_ + bt * 16) * 256ull, 1024);
#else
    {
      const unsigned* src = (const unsigned*)(da.econ + ((size_t)t * B_ + bt * 16) * 128);
      unsigned* dstp = (unsigned*)(ebuf0 + cur * (16 * 128));
      for (int i = tid; i < 16 * 64; i += 128) dstp[i] = src[i];
      __syncthreads();
    }
#endif

    // controller: xp_c = [e|f]@cfWih^T (K=192), hp_c = c@cfWhh^T
#pragma unroll
    for (int q = 0; q < 3; ++q) {
      const int nt = wave * 3 + q;
      v8f xa = {}; v8f ha = {};
#pragma unroll
      for (int ks = 0; ks < 6; ++ks) {
        v16h af = (ks < 4) ? frag_lds(ebuf, lr, 128, ks * 32, lhi)
                           : frag_lds(fbuf, lr, H_, (ks - 4) * 32, lhi);
        v16h bf = frag_lds(cfwih, nt * 16 + lr, G3, ks * 32, lhi);
        xa = wmma32(af, bf, xa);
      }
#pragma unroll
      for (int ks = 0; ks < 2; ++ks) {
        v16h af = frag_lds(c16, lr, H_, ks * 32, lhi);
        v16h bf = frag_lds(cfwhh, nt * 16 + lr, H_, ks * 32, lhi);
        ha = wmma32(af, bf, ha);
      }
      const int n = nt * 16 + lr;
      const float bi = cfbihL[n], bh = cfbhhL[n];
      float* xd = &xpT[n * 16 + lhi * 8];
      float* hd = &hpT[n * 16 + lhi * 8];
#pragma unroll
      for (int r8 = 0; r8 < 8; ++r8) { xd[r8] = xa[r8] + bi; hd[r8] = ha[r8] + bh; }
    }
    __syncthreads();
    for (int i = tid; i < 16 * H_; i += 128) {
      const int m = i >> 6, h = i & 63;
      float rg = sigm(xpT[h * 16 + m] + hpT[h * 16 + m]);
      float zg = sigm(xpT[(64 + h) * 16 + m] + hpT[(64 + h) * 16 + m]);
      float ng = tanhf(xpT[(128 + h) * 16 + m] + rg * hpT[(128 + h) * 16 + m]);
      float cn = clipv((1.f - zg) * ng + zg * c32[i]);
      c32[i] = cn; c16[i] = (_Float16)cn;
    }
    __syncthreads();

    // mu_c / sig_c : 8 tiles, 2 per wave
#pragma unroll
    for (int q = 0; q < 2; ++q) {
      const int tt = wave * 2 + q;
      const _Float16* wm = (tt < 4) ? wmu : wsig;
      const int nb = (tt & 3) * 16;
      v8f acc = {};
#pragma unroll
      for (int ks = 0; ks < 2; ++ks) {
        v16h af = frag_lds(c16, lr, H_, ks * 32, lhi);
        v16h bf = frag_lds(wm, nb + lr, H_, ks * 32, lhi);
        acc = wmma32(af, bf, acc);
      }
      const int n = nb + lr;
      const float bb = (tt < 4) ? bmuL[n] : bsigL[n];
      float* dst = &musigT[((tt < 4 ? 0 : 64) + n) * 16 + lhi * 8];
#pragma unroll
      for (int r8 = 0; r8 < 8; ++r8) dst[r8] = acc[r8] + bb;
    }
    __syncthreads();
    // ut = eps*exp(sig/2)+mu
    for (int i = tid; i < 16 * H_; i += 128) {
      const int m = i >> 6, h = i & 63;
      const int b = bt * 16 + m;
      float ut = da.epsu[((size_t)t * B_ + b) * H_ + h] *
                     expf(musigT[(64 + h) * 16 + m] * 0.5f) +
                 musigT[h * 16 + m];
      u16[i] = (_Float16)ut;
      da.inferred[((size_t)b * H_ + h) * T_ + t] = ut;
    }
    __syncthreads();

    // generator: xp_g = ut@gWih^T, hp_g = g@gWhh^T
#pragma unroll
    for (int q = 0; q < 3; ++q) {
      const int nt = wave * 3 + q;
      v8f xa = {}; v8f ha = {};
#pragma unroll
      for (int ks = 0; ks < 2; ++ks) {
        v16h af = frag_lds(u16, lr, H_, ks * 32, lhi);
        v16h bf = frag_lds(gwih, nt * 16 + lr, H_, ks * 32, lhi);
        xa = wmma32(af, bf, xa);
        v16h af2 = frag_lds(g16, lr, H_, ks * 32, lhi);
        v16h bf2 = frag_lds(gwhh, nt * 16 + lr, H_, ks * 32, lhi);
        ha = wmma32(af2, bf2, ha);
      }
      const int n = nt * 16 + lr;
      const float bi = gbihL[n], bh = gbhhL[n];
      float* xd = &xpT[n * 16 + lhi * 8];
      float* hd = &hpT[n * 16 + lhi * 8];
#pragma unroll
      for (int r8 = 0; r8 < 8; ++r8) { xd[r8] = xa[r8] + bi; hd[r8] = ha[r8] + bh; }
    }
    __syncthreads();
    for (int i = tid; i < 16 * H_; i += 128) {
      const int m = i >> 6, h = i & 63;
      float rg = sigm(xpT[h * 16 + m] + hpT[h * 16 + m]);
      float zg = sigm(xpT[(64 + h) * 16 + m] + hpT[(64 + h) * 16 + m]);
      float ng = tanhf(xpT[(128 + h) * 16 + m] + rg * hpT[(128 + h) * 16 + m]);
      float gn = clipv((1.f - zg) * ng + zg * g32[i]);
      g32[i] = gn; g16[i] = (_Float16)gn;
      da.states[((size_t)(bt * 16 + m) * H_ + h) * T_ + t] = gn;
    }
    __syncthreads();

    // factors: f = g@Wfac^T + b ; write into fbuf + output
    {
      const int nt = wave;  // 4 tiles, one per wave
      v8f acc = {};
#pragma unroll
      for (int ks = 0; ks < 2; ++ks) {
        v16h af = frag_lds(g16, lr, H_, ks * 32, lhi);
        v16h bf = frag_lds(wfac, nt * 16 + lr, H_, ks * 32, lhi);
        acc = wmma32(af, bf, acc);
      }
      const int n = nt * 16 + lr;
      const float bb = bfacL[n];
#pragma unroll
      for (int r8 = 0; r8 < 8; ++r8) {
        const int m = r8 + lhi * 8;
        float fv = acc[r8] + bb;
        fbuf[m * H_ + n] = (_Float16)fv;
        da.factors[((size_t)(bt * 16 + m) * F_ + n) * T_ + t] = fv;
      }
    }
    __syncthreads();

    // rates: r = exp(f@Wrate^T + b) ; 8 tiles, 2 per wave
#pragma unroll
    for (int q = 0; q < 2; ++q) {
      const int nt = wave * 2 + q;
      v8f acc = {};
#pragma unroll
      for (int ks = 0; ks < 2; ++ks) {
        v16h af = frag_lds(fbuf, lr, H_, ks * 32, lhi);
        v16h bf = frag_lds(wrate, nt * 16 + lr, H_, ks * 32, lhi);
        acc = wmma32(af, bf, acc);
      }
      const int n = nt * 16 + lr;
      const float bb = brateL[n];
#pragma unroll
      for (int r8 = 0; r8 < 8; ++r8) {
        const int m = r8 + lhi * 8;
        da.rates[((size_t)(bt * 16 + m) * N_ + n) * T_ + t] = expf(acc[r8] + bb);
      }
    }
    __syncthreads();
  }
}

// --- Loss kernels ------------------------------------------------------------
__global__ void zero_acc_kernel(float* acc) {
  if (threadIdx.x < 4) acc[threadIdx.x] = 0.f;
}

__global__ void ar_kernel(const float* pu, const float* tau_p, const float* pv_p,
                          float* p_u) {
  const int idx = blockIdx.x * blockDim.x + threadIdx.x;  // b*64+h
  const float tau = tau_p[0], pv = pv_p[0];
  const float alpha = expf(-1.f / tau);
  const float eps_std = sqrtf(pv * (1.f - alpha * alpha));
  float u = pu[idx] * pv;
  p_u[(size_t)idx * T_] = u;
  for (int t = 1; t < T_; ++t) {
    u = alpha * u + pu[(size_t)t * B_ * H_ + idx] * eps_std;
    p_u[(size_t)idx * T_ + t] = u;
  }
}

__global__ void poisson_kernel(const float* __restrict__ x,
                               const float* __restrict__ rates, float* acc) {
  __shared__ float red[256];
  const size_t i = (size_t)blockIdx.x * 256 + threadIdx.x;
  float v = 0.f;
  if (i < (size_t)B_ * N_ * T_) {
    float xv = x[i], rv = rates[i];
    v = xv * logf(rv) - rv - lgammaf(xv + 1.f);
  }
  red[threadIdx.x] = v;
  __syncthreads();
  for (int s = 128; s > 0; s >>= 1) {
    if (threadIdx.x < s) red[threadIdx.x] += red[threadIdx.x + s];
    __syncthreads();
  }
  if (threadIdx.x == 0) atomicAdd(acc, red[0]);
}

__device__ float blk64_max(float v, float* sh, int h) {
  sh[h] = v; __syncthreads();
#pragma unroll
  for (int s = 32; s > 0; s >>= 1) {
    if (h < s) sh[h] = fmaxf(sh[h], sh[h + s]);
    __syncthreads();
  }
  float r = sh[0]; __syncthreads();
  return r;
}
__device__ float blk64_sum(float v, float* sh, int h) {
  sh[h] = v; __syncthreads();
#pragma unroll
  for (int s = 32; s > 0; s >>= 1) {
    if (h < s) sh[h] += sh[h + s];
    __syncthreads();
  }
  float r = sh[0]; __syncthreads();
  return r;
}

__global__ void kl_u_kernel(const float* inf_, const float* pu_, float* acc) {
  __shared__ float sh[64];
  const int b = blockIdx.x / T_, t = blockIdx.x - b * T_;
  const int h = threadIdx.x;
  float iv = inf_[((size_t)b * H_ + h) * T_ + t];
  float tv = pu_[((size_t)b * H_ + h) * T_ + t];
  float mi = blk64_max(iv, sh, h);
  float li = iv - mi - logf(blk64_sum(expf(iv - mi), sh, h));
  float mt = blk64_max(tv, sh, h);
  float lt = tv - mt - logf(blk64_sum(expf(tv - mt), sh, h));
  float term = blk64_sum(expf(lt) * (lt - li), sh, h);
  if (h == 0) atomicAdd(acc, term);
}

__global__ void kl_g0_kernel(const float* g0, const float* pg0, float* acc) {
  __shared__ float sh[64];
  const int b = blockIdx.x, h = threadIdx.x;
  float iv = g0[b * H_ + h];
  float tv = pg0[b * H_ + h] * KAPPA;
  float mi = blk64_max(iv, sh, h);
  float li = iv - mi - logf(blk64_sum(expf(iv - mi), sh, h));
  float mt = blk64_max(tv, sh, h);
  float lt = tv - mt - logf(blk64_sum(expf(tv - mt), sh, h));
  float term = blk64_sum(expf(lt) * (lt - li), sh, h);
  if (h == 0) atomicAdd(acc, term);
}

__global__ void finalize_kernel(const float* acc, float* loss) {
  loss[0] = (-acc[0] + (acc[1] + acc[2]) / (float)B_) / (float)B_;
}

// --- Host launcher -----------------------------------------------------------
extern "C" void kernel_launch(void* const* d_in, const int* in_sizes, int n_in,
                              void* d_out, int out_size, void* d_ws,
                              size_t ws_size, hipStream_t stream) {
  (void)in_sizes; (void)n_in; (void)out_size; (void)ws_size;
  const float* x = (const float*)d_in[0];
  const float* a = (const float*)d_in[1];
  auto P = [&](int i) { return (const float*)d_in[i]; };
  const float* eps_g0 = P(42);
  const float* eps_u  = P(43);
  const float* pg0    = P(44);
  const float* pu     = P(45);

  // workspace carve (bytes)
  char* ws = (char*)d_ws;
  _Float16* xa   = (_Float16*)ws;                          // T*B*160 f16
  _Float16* econ = (_Float16*)(ws + 32768000);             // T*B*128 f16
  float* p_u = (float*)(ws + 32768000 + 26214400);         // B*H*T f32
  float* hfw = (float*)(ws + 85196800);
  float* hbw = (float*)(ws + 85262336);
  float* g0  = (float*)(ws + 85327872);
  float* acc = (float*)(ws + 85393408);

  float* out = (float*)d_out;
  float* rates    = out;
  float* factors  = out + (size_t)B_ * N_ * T_;
  float* states   = factors + (size_t)B_ * F_ * T_;
  float* inferred = states + (size_t)B_ * H_ * T_;
  float* loss     = inferred + (size_t)B_ * H_ * T_;

  constexpr int SCAN_LDS = 128512;
  constexpr int DEC_LDS  = 250112;
  hipFuncSetAttribute((const void*)gru_scan_kernel,
                      hipFuncAttributeMaxDynamicSharedMemorySize, SCAN_LDS);
  hipFuncSetAttribute((const void*)dec_kernel,
                      hipFuncAttributeMaxDynamicSharedMemorySize, DEC_LDS);

  zero_acc_kernel<<<1, 64, 0, stream>>>(acc);

  {
    size_t total = (size_t)T_ * B_ * NIP;
    pack_xa_kernel<<<(unsigned)((total + 255) / 256), 256, 0, stream>>>(x, a, xa);
  }

  ScanArgs sa;
  for (int g = 0; g < 4; ++g) {
    sa.Wih[g] = P(2 + 4 * g); sa.Whh[g] = P(3 + 4 * g);
    sa.bih[g] = P(4 + 4 * g); sa.bhh[g] = P(5 + 4 * g);
  }
  sa.xa = xa; sa.econ = econ; sa.hfw = hfw; sa.hbw = hbw;
  gru_scan_kernel<<<dim3(B_ / 16, 4), 128, SCAN_LDS, stream>>>(sa);

  g0_kernel<<<B_, 64, 0, stream>>>(hfw, hbw, P(26), P(27), P(28), P(29), eps_g0, g0);

  DecArgs da;
  da.cfWih = P(18); da.cfWhh = P(19); da.cfbih = P(20); da.cfbhh = P(21);
  da.gWih  = P(22); da.gWhh  = P(23); da.gbih  = P(24); da.gbhh  = P(25);
  da.Wmu = P(34); da.bmu = P(35); da.Wsig = P(36); da.bsig = P(37);
  da.Wfac = P(30); da.bfac = P(31); da.Wrate = P(32); da.brate = P(33);
  da.c0 = P(38); da.f0 = P(39); da.g0 = g0; da.epsu = eps_u; da.econ = econ;
  da.rates = rates; da.factors = factors; da.states = states; da.inferred = inferred;
  dec_kernel<<<B_ / 16, 128, DEC_LDS, stream>>>(da);

  ar_kernel<<<(B_ * H_) / 256, 256, 0, stream>>>(pu, P(40), P(41), p_u);
  poisson_kernel<<<(B_ * N_ * T_) / 256, 256, 0, stream>>>(x, rates, acc);
  kl_u_kernel<<<B_ * T_, 64, 0, stream>>>(inferred, p_u, acc + 1);
  kl_g0_kernel<<<B_, 64, 0, stream>>>(g0, pg0, acc + 2);
  finalize_kernel<<<1, 1, 0, stream>>>(acc, loss);
}
